// CasualSelfAttention_12661563588991
// MI455X (gfx1250) — compile-verified, matrix operand order checked by reading, not run on hardware
//
#include <hip/hip_runtime.h>

// ---------------------------------------------------------------------------
// Causal self-attention on gfx1250 (MI455X): bf16 WMMA GEMMs + flash attention
// Async-LDS staging (ASYNCcnt path) + WMMA 16x16x32 bf16 compute.
// ---------------------------------------------------------------------------

#define BB 4
#define TT 2048
#define CC 1024
#define HH 16
#define DD 64
#define TC (3 * CC)     // 3072
#define MROWS (BB * TT) // 8192

typedef __bf16 bf16_t;
typedef bf16_t v16bf __attribute__((ext_vector_type(16)));
typedef float  v8f   __attribute__((ext_vector_type(8)));
typedef unsigned int v4u __attribute__((ext_vector_type(4)));

union Frag {            // one WMMA 16x32 bf16 operand (8 VGPRs / lane)
    v16bf bf;
    v4u   q[2];         // two 16-byte chunks
};

__device__ __forceinline__ unsigned short f2bf(float f) {
    unsigned int u = __float_as_uint(f);
    unsigned int r = u + 0x7FFFu + ((u >> 16) & 1u);   // round-to-nearest-even
    return (unsigned short)(r >> 16);
}

// ---- CDNA5 async global->LDS copy (ASYNCcnt). LDS offset = low 32 bits of
// the generic shared-memory VA (aperture truncation); global VA in VGPR pair.
__device__ __forceinline__ void async_copy_b128(void* lds_ptr, const void* g_ptr) {
    unsigned loff = (unsigned)(unsigned long long)lds_ptr;
    unsigned long long ga = (unsigned long long)g_ptr;
    asm volatile("global_load_async_to_lds_b128 %0, %1, off"
                 :: "v"(loff), "v"(ga)
                 : "memory");
}
__device__ __forceinline__ void async_wait0() {
    asm volatile("s_wait_asynccnt 0x0" ::: "memory");
}

// Load a 16x32 bf16 A/B fragment: row = lane&15, K chunks at 8*half, 16+8*half.
__device__ __forceinline__ void load_frag(Frag& f, const unsigned short* base, int stride) {
    int lane = threadIdx.x & 31;
    int row = lane & 15, half = lane >> 4;
    const unsigned short* p = base + row * stride;
    f.q[0] = *(const v4u*)(p + 8 * half);
    f.q[1] = *(const v4u*)(p + 16 + 8 * half);
}

// ---------------------------------------------------------------------------
// Kernel 1: f32 -> bf16 cast (grid-stride)
// ---------------------------------------------------------------------------
__global__ __launch_bounds__(256) void cast_f32_bf16_kernel(
    const float* __restrict__ in, unsigned short* __restrict__ out, int n) {
    for (int i = blockIdx.x * 256 + threadIdx.x; i < n; i += gridDim.x * 256)
        out[i] = f2bf(in[i]);
}

// ---------------------------------------------------------------------------
// Kernel 2: W [K][N] f32 -> Wt [N][K] bf16 (32x32 LDS tile transpose)
// ---------------------------------------------------------------------------
__global__ __launch_bounds__(256) void transpose_cast_kernel(
    const float* __restrict__ W, unsigned short* __restrict__ Wt, int K, int N) {
    __shared__ unsigned short tile[32][33];
    int tx = threadIdx.x & 31, ty = threadIdx.x >> 5;   // 32 x 8
    int nb = blockIdx.x * 32, kb = blockIdx.y * 32;
#pragma unroll
    for (int i = ty; i < 32; i += 8)
        tile[i][tx] = f2bf(W[(size_t)(kb + i) * N + nb + tx]);  // tile[k][n]
    __syncthreads();
#pragma unroll
    for (int i = ty; i < 32; i += 8)
        Wt[(size_t)(nb + i) * K + kb + tx] = tile[tx][i];
}

// ---------------------------------------------------------------------------
// Kernel 3/5: bf16 WMMA GEMM: Out[M][N] = A[M][K] * Bt[N][K]^T + bias
// Block tile 128x128, BLK_K=32, 8 waves (2 M-groups x 4 N-groups),
// each wave: 64x32 = 4x2 fragments of 16x16.  Tiles staged via async-LDS.
// ---------------------------------------------------------------------------
template <bool OUT_F32>
__global__ __launch_bounds__(256) void wmma_gemm_kernel(
    const unsigned short* __restrict__ A,   // [M][K] bf16
    const unsigned short* __restrict__ Bt,  // [N][K] bf16
    const float* __restrict__ bias,         // [N]
    void* __restrict__ Out, int M, int N, int K) {
    const int LDT = 48;  // padded row stride (ushorts): 96 B, keeps b128 aligned
    __shared__ unsigned short As[128 * LDT];
    __shared__ unsigned short Bs[128 * LDT];

    int tid = threadIdx.x, lane = tid & 31, wave = tid >> 5;
    int wm = wave & 1, wn = wave >> 1;
    int m0 = blockIdx.y * 128, n0 = blockIdx.x * 128;
    int half = lane >> 4, nl = lane & 15;

    v8f acc[4][2];
#pragma unroll
    for (int i = 0; i < 4; i++)
#pragma unroll
        for (int j = 0; j < 2; j++)
#pragma unroll
            for (int e = 0; e < 8; e++) acc[i][j][e] = 0.0f;

    for (int k0 = 0; k0 < K; k0 += 32) {
        __syncthreads();
        // Stage A tile 128x32 and B tile 128x32 via async global->LDS b128.
#pragma unroll
        for (int u = tid; u < 512; u += 256) {
            int row = u >> 2, c8 = u & 3;
            async_copy_b128(As + row * LDT + c8 * 8,
                            A + (size_t)(m0 + row) * K + k0 + c8 * 8);
        }
#pragma unroll
        for (int u = tid; u < 512; u += 256) {
            int row = u >> 2, c8 = u & 3;
            async_copy_b128(Bs + row * LDT + c8 * 8,
                            Bt + (size_t)(n0 + row) * K + k0 + c8 * 8);
        }
        // Prefetch next K-slab into L2 while WMMAs run on this one.
        if (k0 + 32 < K) {
            int row = tid >> 1, c8 = (tid & 1) * 2;
            __builtin_prefetch(A + (size_t)(m0 + row) * K + k0 + 32 + c8 * 8, 0, 1);
            __builtin_prefetch(Bt + (size_t)(n0 + row) * K + k0 + 32 + c8 * 8, 0, 1);
        }
        async_wait0();
        __syncthreads();

        Frag a[4], b[2];
#pragma unroll
        for (int i = 0; i < 4; i++) load_frag(a[i], As + (wm * 64 + i * 16) * LDT, LDT);
#pragma unroll
        for (int j = 0; j < 2; j++) load_frag(b[j], Bs + (wn * 32 + j * 16) * LDT, LDT);
#pragma unroll
        for (int i = 0; i < 4; i++)
#pragma unroll
            for (int j = 0; j < 2; j++)
                acc[i][j] = __builtin_amdgcn_wmma_f32_16x16x32_bf16(
                    false, a[i].bf, false, b[j].bf, (short)0, acc[i][j], false, false);
    }

    // Epilogue: bias + store. C/D layout: element r -> M = r + 8*half, N = lane&15.
#pragma unroll
    for (int j = 0; j < 2; j++) {
        int n = n0 + wn * 32 + j * 16 + nl;
        float bv = bias ? bias[n] : 0.0f;
#pragma unroll
        for (int i = 0; i < 4; i++) {
            int mb = m0 + wm * 64 + i * 16 + 8 * half;
#pragma unroll
            for (int r = 0; r < 8; r++) {
                float v = acc[i][j][r] + bv;
                size_t idx = (size_t)(mb + r) * N + n;
                if (OUT_F32) ((float*)Out)[idx] = v;
                else ((unsigned short*)Out)[idx] = f2bf(v);
            }
        }
    }
}

// ---------------------------------------------------------------------------
// Kernel 4: flash attention.  qkv bf16 [B*T][3C] -> Y bf16 [B*T][C].
// Block = (b, h, 128 q-rows); 8 waves x 16 q-rows each; K/V tiles of 32 keys.
// ---------------------------------------------------------------------------
__global__ __launch_bounds__(256) void flash_attn_kernel(
    const unsigned short* __restrict__ qkv, unsigned short* __restrict__ Y) {
    const int LDK = 72;  // Ks row stride: 144 B (16B-aligned chunks)
    const int LDV = 48;  // Vs row stride: 96 B
    const int LDP = 48;
    __shared__ unsigned short Ks[32 * LDK];       // [key][d]
    __shared__ unsigned short Vs[64 * LDV];       // [d][key]  (transposed)
    __shared__ unsigned short Ps[8 * 16 * LDP];   // per-wave P staging

    int tid = threadIdx.x, lane = tid & 31, wave = tid >> 5;
    int half = lane >> 4, nl = lane & 15;
    int b = blockIdx.y >> 4, h = blockIdx.y & 15;
    int qbase = blockIdx.x * 128;
    int qw = qbase + wave * 16;

    // Q fragments (A-operand), kept in registers; d chunks [0,32) and [32,64).
    Frag aq[2];
    {
        const unsigned short* qrow = qkv + (size_t)(b * TT + qw + nl) * TC + h * DD;
#pragma unroll
        for (int kc = 0; kc < 2; kc++) {
            aq[kc].q[0] = *(const v4u*)(qrow + 32 * kc + 8 * half);
            aq[kc].q[1] = *(const v4u*)(qrow + 32 * kc + 16 + 8 * half);
        }
    }

    v8f o[4];
    float mstat[8], lstat[8];
#pragma unroll
    for (int nf = 0; nf < 4; nf++)
#pragma unroll
        for (int e = 0; e < 8; e++) o[nf][e] = 0.0f;
#pragma unroll
    for (int r = 0; r < 8; r++) { mstat[r] = -__builtin_inff(); lstat[r] = 0.0f; }

    const int nkt = blockIdx.x * 4 + 4;   // causal: keys [0, qbase+128)
    for (int kt = 0; kt < nkt; kt++) {
        int kb = kt * 32;
        __syncthreads();
        {   // K tile [32][64]: 256 async b128 copies
            int row = tid >> 3, c8 = tid & 7;
            async_copy_b128(Ks + row * LDK + c8 * 8,
                            qkv + (size_t)(b * TT + kb + row) * TC + CC + h * DD + c8 * 8);
        }
        {   // V tile transposed into Vs[d][key] (element permute: manual path)
            int row = tid >> 3, c8 = tid & 7;
            v4u s = *(const v4u*)(qkv + (size_t)(b * TT + kb + row) * TC + 2 * CC + h * DD + c8 * 8);
            const unsigned short* sp = (const unsigned short*)&s;
#pragma unroll
            for (int e = 0; e < 8; e++) Vs[(c8 * 8 + e) * LDV + row] = sp[e];
        }
        async_wait0();
        __syncthreads();

        // S = Q * K^T : 2 d-chunks x 2 key-fragments
        v8f s[2];
#pragma unroll
        for (int j = 0; j < 2; j++)
#pragma unroll
            for (int e = 0; e < 8; e++) s[j][e] = 0.0f;
#pragma unroll
        for (int kc = 0; kc < 2; kc++)
#pragma unroll
            for (int j = 0; j < 2; j++) {
                Frag bk;
                const unsigned short* p = Ks + (j * 16 + nl) * LDK + 32 * kc;
                bk.q[0] = *(const v4u*)(p + 8 * half);
                bk.q[1] = *(const v4u*)(p + 16 + 8 * half);
                s[j] = __builtin_amdgcn_wmma_f32_16x16x32_bf16(
                    false, aq[kc].bf, false, bk.bf, (short)0, s[j], false, false);
            }

        // Online softmax per q-row (row M = r + 8*half lives in one 16-lane half).
        const float scale = 0.125f;  // 1/sqrt(64)
#pragma unroll
        for (int r = 0; r < 8; r++) {
            int qg = qw + r + 8 * half;
            float sv0, sv1;
            {
                int kg0 = kb + nl, kg1 = kb + 16 + nl;
                float v0 = s[0][r] * scale, v1 = s[1][r] * scale;
                sv0 = (kg0 <= qg) ? v0 : -__builtin_inff();
                sv1 = (kg1 <= qg) ? v1 : -__builtin_inff();
            }
            float mx = fmaxf(sv0, sv1);
#pragma unroll
            for (int d = 8; d >= 1; d >>= 1) mx = fmaxf(mx, __shfl_xor(mx, d, 16));
            float mnew = fmaxf(mstat[r], mx);
            float p0 = __expf(sv0 - mnew), p1 = __expf(sv1 - mnew);
            float rs = p0 + p1;
#pragma unroll
            for (int d = 8; d >= 1; d >>= 1) rs += __shfl_xor(rs, d, 16);
            float alpha = __expf(mstat[r] - mnew);
            lstat[r] = lstat[r] * alpha + rs;
            mstat[r] = mnew;
#pragma unroll
            for (int nf = 0; nf < 4; nf++) o[nf][r] *= alpha;
            unsigned short* pp = Ps + (wave * 16 + r + 8 * half) * LDP;
            pp[nl]      = f2bf(p0);
            pp[16 + nl] = f2bf(p1);
        }
        asm volatile("s_wait_dscnt 0" ::: "memory");  // Ps stores -> Ps loads (same wave)

        // O += P * V : P re-enters as A-fragment from per-wave LDS region.
        Frag ap;
        {
            const unsigned short* p = Ps + (wave * 16 + nl) * LDP;
            ap.q[0] = *(const v4u*)(p + 8 * half);
            ap.q[1] = *(const v4u*)(p + 16 + 8 * half);
        }
#pragma unroll
        for (int nf = 0; nf < 4; nf++) {
            Frag bv;
            const unsigned short* p = Vs + (nf * 16 + nl) * LDV;
            bv.q[0] = *(const v4u*)(p + 8 * half);
            bv.q[1] = *(const v4u*)(p + 16 + 8 * half);
            o[nf] = __builtin_amdgcn_wmma_f32_16x16x32_bf16(
                false, ap.bf, false, bv.bf, (short)0, o[nf], false, false);
        }
    }

    // Normalize and store Y.
#pragma unroll
    for (int r = 0; r < 8; r++) {
        float inv = 1.0f / lstat[r];
        int qg = qw + r + 8 * half;
        unsigned short* yrow = Y + (size_t)(b * TT + qg) * CC + h * DD;
#pragma unroll
        for (int nf = 0; nf < 4; nf++) yrow[nf * 16 + nl] = f2bf(o[nf][r] * inv);
    }
}

// ---------------------------------------------------------------------------
extern "C" void kernel_launch(void* const* d_in, const int* in_sizes, int n_in,
                              void* d_out, int out_size, void* d_ws, size_t ws_size,
                              hipStream_t stream) {
    (void)in_sizes; (void)n_in; (void)out_size; (void)ws_size;
    const float* x    = (const float*)d_in[0];
    const float* Wqkv = (const float*)d_in[1];
    const float* bqkv = (const float*)d_in[2];
    const float* Wout = (const float*)d_in[3];
    const float* bout = (const float*)d_in[4];

    size_t off = 0;
    auto seg = [&](size_t bytes) {
        char* p = (char*)d_ws + off;
        off += (bytes + 255) & ~(size_t)255;
        return p;
    };
    unsigned short* xb   = (unsigned short*)seg((size_t)MROWS * CC * 2);  // x bf16
    unsigned short* wqT  = (unsigned short*)seg((size_t)TC * CC * 2);     // Wqkv^T bf16
    unsigned short* woT  = (unsigned short*)seg((size_t)CC * CC * 2);     // Wout^T bf16
    unsigned short* qkvb = (unsigned short*)seg((size_t)MROWS * TC * 2);  // qkv bf16
    unsigned short* yb   = (unsigned short*)seg((size_t)MROWS * CC * 2);  // attn out bf16

    // 1) cast x
    cast_f32_bf16_kernel<<<dim3(8192), 256, 0, stream>>>(x, xb, MROWS * CC);
    // 2) transpose+cast weights to [N][K]
    transpose_cast_kernel<<<dim3(TC / 32, CC / 32), 256, 0, stream>>>(Wqkv, wqT, CC, TC);
    transpose_cast_kernel<<<dim3(CC / 32, CC / 32), 256, 0, stream>>>(Wout, woT, CC, CC);
    // 3) qkv = x @ Wqkv + b   (bf16 out, [8192][3072])
    wmma_gemm_kernel<false><<<dim3(TC / 128, MROWS / 128), 256, 0, stream>>>(
        xb, wqT, bqkv, qkvb, MROWS, TC, CC);
    // 4) flash attention -> y bf16 [8192][1024]
    flash_attn_kernel<<<dim3(TT / 128, BB * HH), 256, 0, stream>>>(qkvb, yb);
    // 5) out = y @ Wout + b   (f32 out)
    wmma_gemm_kernel<true><<<dim3(CC / 128, MROWS / 128), 256, 0, stream>>>(
        yb, woT, bout, d_out, MROWS, CC, CC);
}
